// TextRelationalGraphAttention_78769700209169
// MI455X (gfx1250) — compile-verified, matrix operand order checked by reading, not run on hardware
//
#include <hip/hip_runtime.h>
#include <hip/hip_bf16.h>
#include <math.h>

// N=4096, D=256, O=256, H=256 (2H=512), T=512, S=8, R=4, BN=2
typedef __attribute__((ext_vector_type(16))) _Float16 v16h;
typedef __attribute__((ext_vector_type(8)))  _Float16 v8h;
typedef __attribute__((ext_vector_type(8)))  float    v8f;

// Load a 16-half WMMA fragment whose K pattern is two contiguous 8-half runs.
__device__ __forceinline__ v16h load_frag_h(const _Float16* p0, const _Float16* p1) {
    v8h lo = *(const v8h*)p0;
    v8h hi = *(const v8h*)p1;
    v16h r;
#pragma unroll
    for (int i = 0; i < 8; ++i) { r[i] = lo[i]; r[i + 8] = hi[i]; }
    return r;
}

// ---------------------------------------------------------------------------
// Kernel A: text attention -> context -> c2 -> gated g2   (tiny, 1 block)
// ---------------------------------------------------------------------------
__global__ __launch_bounds__(256) void ctx_kernel(
    const float* __restrict__ te,      // [512][512]  text_encodings[0]
    const int*   __restrict__ adj,     // [4]
    const float* __restrict__ tguids,  // [8][512]
    const float* __restrict__ gg2,     // [8][256]
    const float* __restrict__ guid_a,  // [8]
    const float* __restrict__ W2,      // [512][256]
    float* __restrict__ g2)            // [4][256] (ws)
{
    __shared__ float s_att[512 * 4];
    __shared__ float s_ctx[4 * 512];   // first stages tg, later holds context
    __shared__ float s_red[256];
    const int tid = threadIdx.x;

    // stage tg rows
    for (int i = tid; i < 4 * 512; i += 256) {
        int rr = i >> 9, h = i & 511;
        s_ctx[i] = tguids[adj[rr] * 512 + h];
    }
    __syncthreads();

    // scores[t][r] = te[t] . tg[r]
#pragma unroll
    for (int p = 0; p < 2; ++p) {
        int t = tid * 2 + p;
        const float* ter = te + (size_t)t * 512;
        float d0 = 0.f, d1 = 0.f, d2 = 0.f, d3 = 0.f;
        for (int h = 0; h < 512; ++h) {
            float v = ter[h];
            d0 += v * s_ctx[h];
            d1 += v * s_ctx[512 + h];
            d2 += v * s_ctx[1024 + h];
            d3 += v * s_ctx[1536 + h];
        }
        s_att[t * 4 + 0] = d0; s_att[t * 4 + 1] = d1;
        s_att[t * 4 + 2] = d2; s_att[t * 4 + 3] = d3;
    }
    __syncthreads();

    // softmax over t (axis 0) per relation column
    for (int rr = 0; rr < 4; ++rr) {
        float a0 = s_att[(2 * tid) * 4 + rr];
        float a1 = s_att[(2 * tid + 1) * 4 + rr];
        s_red[tid] = fmaxf(a0, a1);
        __syncthreads();
        for (int s = 128; s > 0; s >>= 1) {
            if (tid < s) s_red[tid] = fmaxf(s_red[tid], s_red[tid + s]);
            __syncthreads();
        }
        float M = s_red[0];
        __syncthreads();
        float e0 = expf(a0 - M), e1 = expf(a1 - M);
        s_red[tid] = e0 + e1;
        __syncthreads();
        for (int s = 128; s > 0; s >>= 1) {
            if (tid < s) s_red[tid] += s_red[tid + s];
            __syncthreads();
        }
        float inv = 1.f / s_red[0];
        __syncthreads();
        s_att[(2 * tid) * 4 + rr] = e0 * inv;
        s_att[(2 * tid + 1) * 4 + rr] = e1 * inv;
        __syncthreads();
    }

    // context[r][h] = sum_t att[t][r] * te[t][h]   (overwrite s_ctx; tg no longer needed)
#pragma unroll
    for (int p = 0; p < 8; ++p) {
        int idx = tid + 256 * p;          // 0..2047
        int rr = idx >> 9, h = idx & 511;
        float sum = 0.f;
        for (int t = 0; t < 512; ++t) sum += s_att[t * 4 + rr] * te[(size_t)t * 512 + h];
        s_ctx[idx] = sum;
    }
    __syncthreads();

    // c2 = context @ W2 ; g2 = ga*graph_guids_2 + (1-ga)*c2
#pragma unroll
    for (int p = 0; p < 4; ++p) {
        int idx = tid + 256 * p;          // 0..1023
        int rr = idx >> 8, d = idx & 255;
        const float* ctx = s_ctx + rr * 512;
        float sum = 0.f;
        for (int h = 0; h < 512; ++h) sum += ctx[h] * W2[(size_t)h * 256 + d];
        float ga = 1.f / (1.f + expf(-guid_a[adj[rr]]));
        g2[idx] = ga * gg2[adj[rr] * 256 + d] + (1.f - ga) * sum;
    }
}

// ---------------------------------------------------------------------------
// Kernel B1: s2[n][r] = E[n] . g2[r]
// ---------------------------------------------------------------------------
__global__ __launch_bounds__(256) void s2_kernel(
    const float* __restrict__ E, const float* __restrict__ g2, float* __restrict__ s2)
{
    __shared__ float sg[1024];
    const int tid = threadIdx.x;
    for (int i = tid; i < 1024; i += 256) sg[i] = g2[i];
    __syncthreads();
    int n = blockIdx.x * 256 + tid;
    const float* er = E + (size_t)n * 256;
    float d0 = 0.f, d1 = 0.f, d2 = 0.f, d3 = 0.f;
    for (int d = 0; d < 256; ++d) {
        float v = er[d];
        d0 += v * sg[d]; d1 += v * sg[256 + d];
        d2 += v * sg[512 + d]; d3 += v * sg[768 + d];
    }
    s2[n * 4 + 0] = d0; s2[n * 4 + 1] = d1;
    s2[n * 4 + 2] = d2; s2[n * 4 + 3] = d3;
}

// ---------------------------------------------------------------------------
// Kernel B2: per-relation global max + w[r][j] = exp(s2[j][r] - M_r)
// ---------------------------------------------------------------------------
__global__ __launch_bounds__(256) void w_kernel(
    const float* __restrict__ s2, float* __restrict__ w)
{
    __shared__ float s_red[256];
    const int tid = threadIdx.x;
    for (int rr = 0; rr < 4; ++rr) {
        float m = -1e30f;
        for (int k = 0; k < 16; ++k) m = fmaxf(m, s2[(tid + 256 * k) * 4 + rr]);
        s_red[tid] = m;
        __syncthreads();
        for (int s = 128; s > 0; s >>= 1) {
            if (tid < s) s_red[tid] = fmaxf(s_red[tid], s_red[tid + s]);
            __syncthreads();
        }
        float M = s_red[0];
        __syncthreads();
        for (int k = 0; k < 16; ++k) {
            int j = tid + 256 * k;
            w[rr * 4096 + j] = expf(s2[j * 4 + rr] - M);
        }
        __syncthreads();
    }
}

// ---------------------------------------------------------------------------
// Kernel C: EwT[r][d][j] = w[r][j] * E[j][d]   (f16, transposed for B frags)
// ---------------------------------------------------------------------------
__global__ __launch_bounds__(256) void ewt_kernel(
    const float* __restrict__ E, const float* __restrict__ w, _Float16* __restrict__ EwT)
{
    const int r = blockIdx.y, d = blockIdx.x;
    const float* wr = w + r * 4096;
    _Float16* outr = EwT + ((size_t)(r * 256 + d)) * 4096;
    for (int j = threadIdx.x; j < 4096; j += 256)
        outr[j] = (_Float16)(wr[j] * E[(size_t)j * 256 + d]);
}

// ---------------------------------------------------------------------------
// Kernel W: WcatT[o][k] = sum_b basis_b[adj[k%4]][b] * basis_V[k/4][b][o]
// (reproduces the reference's (r*D+d) column / (d*R+r) row pairing)
// ---------------------------------------------------------------------------
__global__ __launch_bounds__(256) void wcat_kernel(
    const float* __restrict__ basis_V,  // [256][2][256]
    const float* __restrict__ basis_b,  // [8][2]
    const int*   __restrict__ adj,
    _Float16* __restrict__ WcatT)       // [256][1024]
{
    const int k = blockIdx.x;           // 0..1023
    const int o = threadIdx.x;          // 0..255
    const int dp = k >> 2, rp = k & 3;
    float b0 = basis_b[adj[rp] * 2 + 0];
    float b1 = basis_b[adj[rp] * 2 + 1];
    float v = b0 * basis_V[(size_t)(dp * 2 + 0) * 256 + o]
            + b1 * basis_V[(size_t)(dp * 2 + 1) * 256 + o];
    WcatT[(size_t)o * 1024 + k] = (_Float16)v;
}

// ---------------------------------------------------------------------------
// Kernel D: fused[r] = rownorm( Mask_r @ (w_r ⊙ E) )   -- the 34 GFLOP GEMM.
// Wave = 16 rows x full D=256 (16 WMMA tiles); mask read exactly once,
// converted to f16 A-fragments in-register; Z accumulated from same stream.
// ---------------------------------------------------------------------------
__global__ __launch_bounds__(256) void fused_attn_gemm(
    const int* __restrict__ mask,       // [4][4096][4096]
    const float* __restrict__ w,        // [4][4096]
    const _Float16* __restrict__ EwT,   // [4][256][4096]
    _Float16* __restrict__ fusedcat)    // [4096][1024], col = r*256+d
{
    const int r    = blockIdx.y;
    const int wave = threadIdx.x >> 5;
    const int lane = threadIdx.x & 31;
    const int g    = lane >> 4;         // K-subset group per ISA 16-bit A layout
    const int lrow = lane & 15;         // matrix row within tile
    const int i0   = blockIdx.x * 128 + wave * 16;
    const int row  = i0 + lrow;

    const int*      mrow = mask + ((size_t)r * 4096 + row) * 4096;
    const float*    wr   = w + r * 4096;
    const _Float16* Er   = EwT + (size_t)r * 256 * 4096;

    v8f acc[16];
#pragma unroll
    for (int t = 0; t < 16; ++t)
#pragma unroll
        for (int e = 0; e < 8; ++e) acc[t][e] = 0.f;
    float zacc = 0.f;

    for (int j0 = 0; j0 < 4096; j0 += 32) {
        const int ja = j0 + 8 * g;
        const int jb = ja + 16;
        // prefetch mask ahead (gfx1250 global_prefetch_b8)
        if (j0 + 256 < 4096) __builtin_prefetch(mrow + j0 + 256, 0, 1);

        int4 m0 = *(const int4*)(mrow + ja);
        int4 m1 = *(const int4*)(mrow + ja + 4);
        int4 m2 = *(const int4*)(mrow + jb);
        int4 m3 = *(const int4*)(mrow + jb + 4);
        float4 w0 = *(const float4*)(wr + ja);
        float4 w1 = *(const float4*)(wr + ja + 4);
        float4 w2 = *(const float4*)(wr + jb);
        float4 w3 = *(const float4*)(wr + jb + 4);

        int   mv[16] = { m0.x, m0.y, m0.z, m0.w, m1.x, m1.y, m1.z, m1.w,
                         m2.x, m2.y, m2.z, m2.w, m3.x, m3.y, m3.z, m3.w };
        float wv[16] = { w0.x, w0.y, w0.z, w0.w, w1.x, w1.y, w1.z, w1.w,
                         w2.x, w2.y, w2.z, w2.w, w3.x, w3.y, w3.z, w3.w };
        v16h a;
#pragma unroll
        for (int i = 0; i < 16; ++i) {
            bool on = (mv[i] == 1);
            a[i] = on ? (_Float16)1.0f : (_Float16)0.0f;
            zacc += on ? wv[i] : 0.f;
        }

#pragma unroll
        for (int dt = 0; dt < 16; ++dt) {
            const _Float16* bp = Er + (size_t)(dt * 16 + lrow) * 4096 + ja;
            v16h b = load_frag_h(bp, bp + 16);
            acc[dt] = __builtin_amdgcn_wmma_f32_16x16x32_f16(
                false, a, false, b, (short)0, acc[dt], false, false);
        }
    }

    // Z reduction: lanes L and L+16 cover complementary K halves of row L%16.
    zacc += __shfl_xor(zacc, 16, 32);
    float rinv = 1.0f / zacc;
    float rz[8];
#pragma unroll
    for (int e = 0; e < 8; ++e) rz[e] = __shfl(rinv, (e + 8 * g) + 16 * g * 0 + 0 * 0 + 0, 32);
    // (src lane e+8g holds rinv for row e+8g; both half-groups have identical values)

#pragma unroll
    for (int dt = 0; dt < 16; ++dt) {
#pragma unroll
        for (int e = 0; e < 8; ++e) {
            int n = i0 + e + 8 * g;
            int col = r * 256 + dt * 16 + lrow;
            fusedcat[(size_t)n * 1024 + col] = (_Float16)(acc[dt][e] * rz[e]);
        }
    }
}

// ---------------------------------------------------------------------------
// Kernel E: out = LN(fused_cat @ Wcat) -> ReLU     (WMMA f16, LN in registers)
// ---------------------------------------------------------------------------
__global__ __launch_bounds__(256) void final_gemm_ln(
    const _Float16* __restrict__ fusedcat, // [4096][1024]
    const _Float16* __restrict__ WcatT,    // [256][1024]
    const float* __restrict__ gamma,
    const float* __restrict__ beta,
    float* __restrict__ out)               // [4096][256]
{
    const int wave = threadIdx.x >> 5;
    const int lane = threadIdx.x & 31;
    const int g    = lane >> 4;
    const int lrow = lane & 15;
    const int n0   = blockIdx.x * 128 + wave * 16;

    const _Float16* arow = fusedcat + (size_t)(n0 + lrow) * 1024;

    v8f acc[16];
#pragma unroll
    for (int t = 0; t < 16; ++t)
#pragma unroll
        for (int e = 0; e < 8; ++e) acc[t][e] = 0.f;

    for (int j0 = 0; j0 < 1024; j0 += 32) {
        const int ja = j0 + 8 * g;
        v16h a = load_frag_h(arow + ja, arow + ja + 16);
#pragma unroll
        for (int ot = 0; ot < 16; ++ot) {
            const _Float16* bp = WcatT + (size_t)(ot * 16 + lrow) * 1024 + ja;
            v16h b = load_frag_h(bp, bp + 16);
            acc[ot] = __builtin_amdgcn_wmma_f32_16x16x32_f16(
                false, a, false, b, (short)0, acc[ot], false, false);
        }
    }

    float gam[16], bet[16];
#pragma unroll
    for (int ot = 0; ot < 16; ++ot) {
        gam[ot] = gamma[ot * 16 + lrow];
        bet[ot] = beta[ot * 16 + lrow];
    }

#pragma unroll
    for (int e = 0; e < 8; ++e) {
        float s = 0.f, q = 0.f;
#pragma unroll
        for (int ot = 0; ot < 16; ++ot) { float v = acc[ot][e]; s += v; q += v * v; }
        // reduce across the 16 lanes of this half-group (row = e + 8g)
#pragma unroll
        for (int off = 1; off < 16; off <<= 1) {
            s += __shfl_xor(s, off, 32);
            q += __shfl_xor(q, off, 32);
        }
        float mean = s * (1.f / 256.f);
        float var  = q * (1.f / 256.f) - mean * mean;
        float rstd = rsqrtf(var + 1e-5f);
        int n = n0 + e + 8 * g;
#pragma unroll
        for (int ot = 0; ot < 16; ++ot) {
            int o = ot * 16 + lrow;
            float v = (acc[ot][e] - mean) * rstd * gam[ot] + bet[ot];
            out[(size_t)n * 256 + o] = v > 0.f ? v : 0.f;
        }
    }
}

// ---------------------------------------------------------------------------
extern "C" void kernel_launch(void* const* d_in, const int* in_sizes, int n_in,
                              void* d_out, int out_size, void* d_ws, size_t ws_size,
                              hipStream_t stream) {
    const float* E       = (const float*)d_in[0];   // [4096][256]
    const float* te      = (const float*)d_in[1];   // [1][512][512]
    const int*   adj     = (const int*)  d_in[2];   // [4]
    const int*   masks   = (const int*)  d_in[3];   // [4][4096][4096]
    const float* tguids  = (const float*)d_in[4];   // [8][512]
    // d_in[5] graph_guids_1: unused (s1 cancels in the row softmax)
    const float* gg2     = (const float*)d_in[6];   // [8][256]
    const float* guid_a  = (const float*)d_in[7];   // [8]
    const float* basis_V = (const float*)d_in[8];   // [256][2][256]
    const float* basis_b = (const float*)d_in[9];   // [8][2]
    // d_in[10] W_1: unused
    const float* W2      = (const float*)d_in[11];  // [512][256]
    const float* gamma   = (const float*)d_in[12];  // [256]
    const float* beta    = (const float*)d_in[13];  // [256]
    float* out = (float*)d_out;                     // [4096][256]

    char* ws = (char*)d_ws;
    float*    g2       = (float*)   (ws + 0);          //  4 KB
    float*    s2       = (float*)   (ws + 4096);       // 64 KB
    float*    wbuf     = (float*)   (ws + 69632);      // 64 KB
    _Float16* EwT      = (_Float16*)(ws + 135168);     //  8 MB
    _Float16* fusedcat = (_Float16*)(ws + 8523776);    //  8 MB
    _Float16* WcatT    = (_Float16*)(ws + 16912384);   // 512 KB

    ctx_kernel<<<1, 256, 0, stream>>>(te, adj, tguids, gg2, guid_a, W2, g2);
    s2_kernel<<<16, 256, 0, stream>>>(E, g2, s2);
    w_kernel<<<1, 256, 0, stream>>>(s2, wbuf);
    ewt_kernel<<<dim3(256, 4), 256, 0, stream>>>(E, wbuf, EwT);
    wcat_kernel<<<1024, 256, 0, stream>>>(basis_V, basis_b, adj, WcatT);
    fused_attn_gemm<<<dim3(32, 4), 256, 0, stream>>>(masks, wbuf, EwT, fusedcat);
    final_gemm_ln<<<32, 256, 0, stream>>>(fusedcat, WcatT, gamma, beta, out);
}
